// SelfAttnLayer_3770981286198
// MI455X (gfx1250) — matrix operand order repaired and checked by reading, not compiled, not run on hardware
//
#include <hip/hip_runtime.h>
#include <hip/hip_bf16.h>

// ---- problem constants (match reference) ----
constexpr int BS  = 2;
constexpr int SEQ = 2048;
constexpr int DM  = 1024;   // INPUT_DIM
constexpr int H   = 16;     // NUM_HEADS
constexpr int DH  = 64;     // head dim
constexpr int KT  = 32;     // key tile (streamed)
constexpr int QW  = 4;      // waves (16-query tiles) per attention block

typedef __attribute__((ext_vector_type(16))) _Float16 v16h;
typedef __attribute__((ext_vector_type(8)))  float    v8f;

// D = A(16x32 f16) * B(32x16 f16) + C(16x16 f32)
__device__ inline v8f wmma_f16(v16h a, v16h b, v8f c) {
  return __builtin_amdgcn_wmma_f32_16x16x32_f16(
      /*neg_a=*/false, a, /*neg_b=*/false, b,
      /*c_mod=*/(short)0, c, /*reuse_a=*/false, /*reuse_b=*/false);
}

// A-matrix 16x32 f16 fragment from row-major [*, ld].
// lanes 0-15 -> M=lane, K 0..15; lanes 16-31 -> K 16..31 (contiguous).
__device__ inline v16h load_a16(const _Float16* base, int ld, int lane) {
  const int m  = lane & 15;
  const int kb = (lane & 16) ? 16 : 0;
  const _Float16* p = base + (size_t)m * ld + kb;
  v16h r;
#pragma unroll
  for (int i = 0; i < 16; ++i) r[i] = p[i];
  return r;
}

// B-matrix 32x16 f16 fragment where B = M^T of row-major M[*, ld]:
// B[k][n] = M[n][k]  (contiguous 16-half read per lane).
// Contiguous k (same as load_a16): lanes 0-15 hold K=0..15, lanes 16-31 hold K=16..31; N = lane&15.
__device__ inline v16h load_bT16(const _Float16* base, int ld, int lane) {
  const int n  = lane & 15;
  const int kb = lane & 16;     // 0 or 16
  const _Float16* p = base + (size_t)n * ld + kb;
  v16h r;
#pragma unroll
  for (int i = 0; i < 16; ++i) r[i] = p[i];
  return r;
}

// ---- CDNA5 async global->LDS copy (one b128 per lane, ASYNCcnt-tracked) ----
__device__ inline void async_copy_b128(unsigned lds_byte_off, const void* gaddr) {
  asm volatile("global_load_async_to_lds_b128 %0, %1, off"
               :: "v"(lds_byte_off), "v"((unsigned long long)(uintptr_t)gaddr)
               : "memory");
}
__device__ inline void wait_async_le(int) = delete;
__device__ inline void wait_async_4() { asm volatile("s_wait_asynccnt 0x4" ::: "memory"); }
__device__ inline void wait_async_0() { asm volatile("s_wait_asynccnt 0x0" ::: "memory"); }

// low 32 bits of a generic pointer to LDS == byte offset within the WG's LDS
__device__ inline unsigned lds_off(const void* p) {
  return (unsigned)(uintptr_t)p;
}

// ---------------- elementwise f32 -> f16 conversion ----------------
__global__ void cvt_f32_f16(const float* __restrict__ src,
                            _Float16* __restrict__ dst, int n) {
  int i = blockIdx.x * blockDim.x + threadIdx.x;
  const int stride = gridDim.x * blockDim.x;
  for (; i < n; i += stride) dst[i] = (_Float16)src[i];
}

// ---------------- fused QKV projection ----------------
// One wave computes a 16x16 tile of (X @ W^T + b).
// Q, K stored f16 as [b, h, seq, d];  V stored TRANSPOSED as [b, h, d, seq]
// so the attention P@V B-operand becomes a contiguous per-lane read.
__global__ __launch_bounds__(32) void proj_kernel(
    const _Float16* __restrict__ X,
    const _Float16* __restrict__ Wq, const _Float16* __restrict__ Wk,
    const _Float16* __restrict__ Wv,
    const float* __restrict__ bq, const float* __restrict__ bk,
    const float* __restrict__ bv,
    _Float16* __restrict__ Qo, _Float16* __restrict__ Ko,
    _Float16* __restrict__ Vo) {
  const int lane = threadIdx.x;
  const int m0 = blockIdx.x * 16;          // global row in [0, BS*SEQ)
  const int n0 = blockIdx.y * 16;          // output feature in [0, DM)
  const int which = blockIdx.z;

  const _Float16* W = (which == 0) ? Wq : (which == 1) ? Wk : Wv;
  const float* bias = (which == 0) ? bq : (which == 1) ? bk : bv;
  _Float16* dst     = (which == 0) ? Qo : (which == 1) ? Ko : Vo;
  const bool vtrans = (which == 2);

  v8f c = {};
  for (int k0 = 0; k0 < DM; k0 += 32) {
    v16h a  = load_a16(X + (size_t)m0 * DM + k0, DM, lane);
    v16h bf = load_bT16(W + (size_t)n0 * DM + k0, DM, lane);  // B = W^T
    c = wmma_f16(a, bf, c);
  }

  const int nl   = lane & 15;
  const int mrow = (lane >> 4) << 3;       // 0 or 8
  const int n    = n0 + nl;
  const float bb = bias[n];
  const int hh   = n >> 6;                 // head
  const int dd   = n & 63;                 // within-head dim
#pragma unroll
  for (int r = 0; r < 8; ++r) {
    const int M    = m0 + mrow + r;
    const int bidx = M >> 11;              // / SEQ
    const int s    = M & 2047;             // % SEQ
    const _Float16 val = (_Float16)(c[r] + bb);
    if (vtrans)
      dst[(((size_t)(bidx * H + hh) * DH) + dd) * SEQ + s] = val;   // [b,h,d,s]
    else
      dst[(((size_t)(bidx * H + hh) * SEQ) + s) * DH + dd] = val;   // [b,h,s,d]
  }
}

// ---------------- flash attention with async-LDS staged K/V ----------------
// 4 waves per block (64 queries) share double-buffered K/V tiles staged via
// global_load_async_to_lds_b128; per-wave online softmax in registers.
__global__ __launch_bounds__(128) void attn_kernel(
    const _Float16* __restrict__ Q, const _Float16* __restrict__ K,
    const _Float16* __restrict__ Vt, const int* __restrict__ mask,
    float* __restrict__ out) {
  __shared__ _Float16 Ks[2][KT * DH];      // [key][d]    4 KB each
  __shared__ _Float16 Vs[2][DH * KT];      // [d][key]    4 KB each
  __shared__ _Float16 pbuf[QW][16 * 32];   // per-wave P relayout, 1 KB each

  const int tid  = threadIdx.x;
  const int wave = tid >> 5;
  const int lane = tid & 31;
  const int h = blockIdx.y;
  const int b = blockIdx.z;

  const size_t headoff = (size_t)(b * H + h) * SEQ * DH;
  const _Float16* Qb  = Q  + headoff;
  const _Float16* Kb  = K  + headoff;
  const _Float16* Vtb = Vt + headoff;      // [d][seq] within this head
  const int* mb = mask + b * SEQ;

  const int q0   = (blockIdx.x * QW + wave) * 16;
  const int nl   = lane & 15;
  const int mrow = (lane >> 4) << 3;       // 0 or 8

  // stage one 32-key tile: K tile is one contiguous 4 KB run; V^T tile is
  // 64 rows x 64 B. 256 b128 chunks each, 2 per thread per tile.
  auto stage = [&](int buf, int k0) {
    const char* gk = (const char*)(Kb + (size_t)k0 * DH);
    const unsigned lk = lds_off(&Ks[buf][0]);
    async_copy_b128(lk + tid * 16,         gk + tid * 16);
    async_copy_b128(lk + (tid + 128) * 16, gk + (tid + 128) * 16);
    const unsigned lv = lds_off(&Vs[buf][0]);
    {
      const int c = tid, row = c >> 2, sub = c & 3;
      const char* gv = (const char*)(Vtb + (size_t)row * SEQ + k0) + sub * 16;
      async_copy_b128(lv + c * 16, gv);
    }
    {
      const int c = tid + 128, row = c >> 2, sub = c & 3;
      const char* gv = (const char*)(Vtb + (size_t)row * SEQ + k0) + sub * 16;
      async_copy_b128(lv + c * 16, gv);
    }
  };

  // Q tile A-fragments (two K=32 steps over d=64), kept in registers
  const v16h qa0 = load_a16(Qb + (size_t)q0 * DH,      DH, lane);
  const v16h qa1 = load_a16(Qb + (size_t)q0 * DH + 32, DH, lane);

  v8f o0 = {}, o1 = {}, o2 = {}, o3 = {};  // 16x64 f32 accumulator
  float rm[8], rl[8];
#pragma unroll
  for (int r = 0; r < 8; ++r) { rm[r] = -3.0e38f; rl[r] = 0.0f; }

  const float inv_sqrt_d = 0.125f;         // 1/sqrt(64)

  stage(0, 0);                             // prologue: prefetch tile 0
  int buf = 0;

  for (int k0 = 0; k0 < SEQ; k0 += KT) {
    // issue next tile, then wait for the current one (in-order completion)
    if (k0 + KT < SEQ) { stage(buf ^ 1, k0 + KT); wait_async_4(); }
    else               { wait_async_0(); }
    __syncthreads();                       // staged data visible to all waves

    const _Float16* Kt = &Ks[buf][0];
    const _Float16* Vp = &Vs[buf][0];

    // ---- S = Q @ K^T : two 16x16 key sub-tiles, K-contraction = 64 ----
    v8f s0 = {}, s1 = {};
    s0 = wmma_f16(qa0, load_bT16(Kt,                DH, lane), s0);
    s0 = wmma_f16(qa1, load_bT16(Kt + 32,           DH, lane), s0);
    s1 = wmma_f16(qa0, load_bT16(Kt + 16 * DH,      DH, lane), s1);
    s1 = wmma_f16(qa1, load_bT16(Kt + 16 * DH + 32, DH, lane), s1);

    // ---- mask (before scaling, like reference) + scale ----
    const float mv0 = (float)mb[k0 + nl];
    const float mv1 = (float)mb[k0 + 16 + nl];
    const float pen0 = (1.0f - mv0) * (-1.0e30f);
    const float pen1 = (1.0f - mv1) * (-1.0e30f);

    float p0[8], p1[8], tmax[8];
#pragma unroll
    for (int r = 0; r < 8; ++r) {
      p0[r] = (s0[r] * mv0 + pen0) * inv_sqrt_d;
      p1[r] = (s1[r] * mv1 + pen1) * inv_sqrt_d;
      tmax[r] = fmaxf(p0[r], p1[r]);
    }
    // row max: a softmax row lives in one 16-lane group in the C/D layout
#pragma unroll
    for (int r = 0; r < 8; ++r) {
#pragma unroll
      for (int off = 1; off < 16; off <<= 1)
        tmax[r] = fmaxf(tmax[r], __shfl_xor(tmax[r], off, 32));
    }
    float corr[8], tsum[8];
#pragma unroll
    for (int r = 0; r < 8; ++r) {
      const float mnew = fmaxf(rm[r], tmax[r]);
      corr[r] = __expf(rm[r] - mnew);
      rm[r] = mnew;
      p0[r] = __expf(p0[r] - mnew);
      p1[r] = __expf(p1[r] - mnew);
      tsum[r] = p0[r] + p1[r];
    }
#pragma unroll
    for (int r = 0; r < 8; ++r) {
#pragma unroll
      for (int off = 1; off < 16; off <<= 1)
        tsum[r] += __shfl_xor(tsum[r], off, 32);
      rl[r] = rl[r] * corr[r] + tsum[r];
      o0[r] *= corr[r];
      o1[r] *= corr[r];
      o2[r] *= corr[r];
      o3[r] *= corr[r];
    }

    // ---- re-layout P (D fragment -> A fragment) via per-wave LDS slice ----
    _Float16* pw = &pbuf[wave][0];
#pragma unroll
    for (int r = 0; r < 8; ++r) {
      pw[(mrow + r) * 32 + nl]      = (_Float16)p0[r];
      pw[(mrow + r) * 32 + 16 + nl] = (_Float16)p1[r];
    }
    const v16h pa = load_a16(pw, 32, lane);

    // ---- O += P @ V : V^T tile gives contiguous per-lane B reads ----
    o0 = wmma_f16(pa, load_bT16(Vp +  0 * KT, KT, lane), o0);
    o1 = wmma_f16(pa, load_bT16(Vp + 16 * KT, KT, lane), o1);
    o2 = wmma_f16(pa, load_bT16(Vp + 32 * KT, KT, lane), o2);
    o3 = wmma_f16(pa, load_bT16(Vp + 48 * KT, KT, lane), o3);

    __syncthreads();                       // all reads done before overwrite
    buf ^= 1;
  }

  // ---- normalize + store (output layout [b, s, h*d], f32) ----
#pragma unroll
  for (int r = 0; r < 8; ++r) {
    const float invl = 1.0f / rl[r];
    const int m = q0 + mrow + r;
    float* op = out + ((size_t)b * SEQ + m) * DM + h * DH + nl;
    op[0]  = o0[r] * invl;
    op[16] = o1[r] * invl;
    op[32] = o2[r] * invl;
    op[48] = o3[r] * invl;
  }
}

extern "C" void kernel_launch(void* const* d_in, const int* in_sizes, int n_in,
                              void* d_out, int out_size, void* d_ws, size_t ws_size,
                              hipStream_t stream) {
  const float* x    = (const float*)d_in[0];
  const int*   mask = (const int*)d_in[1];
  const float* Wq   = (const float*)d_in[2];
  const float* bq   = (const float*)d_in[3];
  const float* Wk   = (const float*)d_in[4];
  const float* bk   = (const float*)d_in[5];
  const float* Wv   = (const float*)d_in[6];
  const float* bv   = (const float*)d_in[7];
  float* out = (float*)d_out;

  // workspace carve-up (f16 elements): ~38 MB total
  _Float16* p   = (_Float16*)d_ws;
  _Float16* X16 = p;  p += (size_t)BS * SEQ * DM;
  _Float16* Wq16 = p; p += (size_t)DM * DM;
  _Float16* Wk16 = p; p += (size_t)DM * DM;
  _Float16* Wv16 = p; p += (size_t)DM * DM;
  _Float16* Q16 = p;  p += (size_t)BS * SEQ * DM;
  _Float16* K16 = p;  p += (size_t)BS * SEQ * DM;
  _Float16* V16 = p;  p += (size_t)BS * SEQ * DM;   // stored [b,h,d,seq]

  cvt_f32_f16<<<2048, 256, 0, stream>>>(x,  X16,  BS * SEQ * DM);
  cvt_f32_f16<<<1024, 256, 0, stream>>>(Wq, Wq16, DM * DM);
  cvt_f32_f16<<<1024, 256, 0, stream>>>(Wk, Wk16, DM * DM);
  cvt_f32_f16<<<1024, 256, 0, stream>>>(Wv, Wv16, DM * DM);

  dim3 pg(BS * SEQ / 16, DM / 16, 3);   // 256 x 64 x 3 waves
  proj_kernel<<<pg, 32, 0, stream>>>(X16, Wq16, Wk16, Wv16, bq, bk, bv,
                                     Q16, K16, V16);

  dim3 ag(SEQ / (16 * QW), H, BS);      // 32 x 16 x 2 blocks of 4 waves
  attn_kernel<<<ag, 128, 0, stream>>>(Q16, K16, V16, mask, out);
}
